// QuantumMultiheadAttention_9113920602187
// MI455X (gfx1250) — matrix operand order repaired and checked by reading, not compiled, not run on hardware
//
#include <hip/hip_runtime.h>
#include <hip/hip_bf16.h>
#include <math.h>

// Shapes fixed by the reference.
#define B_    8
#define NQ    1024
#define NK    4096
#define DD    64
#define SS    32          // NUM_SAMPLES
#define QT    16          // queries per workgroup (WMMA N tile)
#define NTILES (NK / 16)  // 256 key tiles of 16
#define NWAVES 4
#define SCALE 0.125f      // 1/sqrt(64)

typedef __attribute__((ext_vector_type(16))) __bf16          v16bf;
typedef __attribute__((ext_vector_type(8)))  float           v8f;
typedef __attribute__((ext_vector_type(4)))  unsigned int    v4u;
typedef __attribute__((ext_vector_type(8)))  int             v8i;
typedef __attribute__((ext_vector_type(4)))  int             v4i;

#if __has_builtin(__builtin_amdgcn_tensor_load_to_lds)
#define USE_TDM 1
// Staged K tile in LDS: D# padding of 4 dwords every 64 dwords -> row stride
// 68 dwords, so lane n's row base sits at bank (68*n)%64 = 4n: the 16 lanes of
// a half-wave hit 16 distinct bank groups -> conflict-free ds_load_b128.
#define KSTRIDE 68
#define KSTAGE_FLOATS (16 * KSTRIDE + 4)
#endif

__device__ __forceinline__ unsigned hash32(unsigned x) {
  x ^= x >> 16; x *= 0x7feb352dU;
  x ^= x >> 15; x *= 0x846ca68bU;
  x ^= x >> 16;
  return x;
}

// Exchange with lane^16 (cross half-wave) as pure VALU — no LDS round-trip.
__device__ __forceinline__ float xhalf(float v) {
#if __has_builtin(__builtin_amdgcn_permlanex16)
  int i = __builtin_amdgcn_permlanex16(__float_as_int(v), __float_as_int(v),
                                       0x76543210, 0xfedcba98, false, false);
  return __int_as_float(i);
#else
  return __shfl_xor(v, 16, 32);
#endif
}

#ifdef USE_TDM
// One TDM descriptor: 2D tile, 16 rows x 64 f32, row stride 64 elements in
// memory; LDS padding 4 dwords per 64-dword interval (see KSTRIDE).
__device__ __forceinline__ void tdm_load_ktile(const float* gaddr,
                                               unsigned lds_off) {
  const unsigned long long ga = (unsigned long long)(size_t)gaddr;
  v4u g0;
  g0[0] = 1u;                                   // count=1, user descriptor
  g0[1] = lds_off;                              // lds_addr
  g0[2] = (unsigned)(ga & 0xFFFFFFFFull);       // global_addr[31:0]
  g0[3] = (unsigned)((ga >> 32) & 0x1FFFFFFull) // global_addr[56:32]
          | (2u << 30);                         // type = 2 ("image")
  v8i g1;
  g1[0] = (int)(0x00020000u                     // data_size = 4B
                | (1u << 20)                    // pad_enable
                | (5u << 22)                    // pad_interval: 64 dwords
                | (3u << 25));                  // pad_amount:   4 dwords
  g1[1] = (int)(64u << 16);                     // tensor_dim0 = 64
  g1[2] = (int)(4096u << 16);                   // tensor_dim1 = 4096
  g1[3] = (int)(64u << 16);                     // tile_dim0 = 64
  g1[4] = 16;                                   // tile_dim1 = 16
  g1[5] = 64;                                   // tensor_dim0_stride = 64
  g1[6] = 0;
  g1[7] = 0;
  const v4i gz = {0, 0, 0, 0};
#if __clang_major__ >= 23
  const v8i gz8 = {0, 0, 0, 0, 0, 0, 0, 0};
  __builtin_amdgcn_tensor_load_to_lds(g0, g1, gz, gz, gz8, 0);
#else
  __builtin_amdgcn_tensor_load_to_lds(g0, g1, gz, gz, 0);
#endif
}
#endif

__global__ __launch_bounds__(128)
void qattn_sample_kernel(const float* __restrict__ Q,
                         const float* __restrict__ K,
                         const float* __restrict__ V,
                         float* __restrict__ Out) {
  __shared__ float Qs[QT * DD];                 // 4 KB, f32 Q tile
  __shared__ float tileMax[QT][NTILES];         // 16 KB  [query][keytile]
  __shared__ float tileSum[QT][NTILES];         // 16 KB (rescaled in place)
  __shared__ float wpM[NWAVES][QT];
  __shared__ float wpL[NWAVES][QT];
  __shared__ float rowM[QT];
  __shared__ float rowL[QT];
  __shared__ unsigned short sampKey[QT][SS];    // 1 KB
#ifdef USE_TDM
  __shared__ float Kstage[NWAVES][2][KSTAGE_FLOATS];  // ~35 KB, TDM staging
#endif

  const int tid  = threadIdx.x;
  const int wave = tid >> 5;
  const int lane = tid & 31;
  const int n    = lane & 15;   // query column (C layout N) / fragment row
  const int h    = lane >> 4;   // half-wave
  const int blk  = blockIdx.x;
  const int b    = blk >> 6;          // blk / (NQ/QT)
  const int q0   = (blk & 63) * QT;

  const float* Kb = K + (size_t)b * NK * DD;
  const float* Vb = V + (size_t)b * NK * DD;

  // ---- stage Q tile (16 x 64 f32) into LDS ----
  for (int i = tid; i < QT * DD; i += 128)
    Qs[i] = Q[(size_t)(b * NQ + q0) * DD + i];
  __syncthreads();

  // ---- stationary bf16 B fragments: B[kk][n] = Q[q0+n][slice*32+kk]
  //      (32x16 B layout: lane n = column, element j -> kk = (j>=8 ? 16 : 0) + h*8 + (j&7)) ----
  v16bf bq0, bq1;
  #pragma unroll
  for (int j = 0; j < 16; ++j) {
    const int kk = ((j >> 3) << 4) + (h << 3) + (j & 7);  // 0..31 within slice
    bq0[j] = (__bf16)Qs[n * DD + kk];
    bq1[j] = (__bf16)Qs[n * DD + 32 + kk];
  }

  // ---- pass 1: stream K tiles; logits^T tile = K_tile x Q^T via WMMA ----
  // C layout: lane holds query n, keys M = r + 8h (r = accumulator index).
  float m_run = -INFINITY, l_run = 0.0f;  // flash stats for query n

#ifdef USE_TDM
  const unsigned stage0 = (unsigned)(size_t)&Kstage[wave][0][0];
  const unsigned stage1 = (unsigned)(size_t)&Kstage[wave][1][0];
  tdm_load_ktile(Kb + (size_t)wave * 16 * DD, stage0);  // prologue: tile #wave
  int par = 0;
#endif

  for (int t = wave; t < NTILES; t += NWAVES) {
#ifdef USE_TDM
    // Double-buffered TDM pipeline: issue next tile's DMA, then wait until
    // only that newest op is outstanding (in-order => current tile is ready).
    if (t + NWAVES < NTILES) {
      tdm_load_ktile(Kb + (size_t)(t + NWAVES) * 16 * DD,
                     par ? stage0 : stage1);
      __builtin_amdgcn_s_wait_tensorcnt((short)1);
    } else {
      __builtin_amdgcn_s_wait_tensorcnt((short)0);
    }
    const float* krow =
        (par ? &Kstage[wave][1][0] : &Kstage[wave][0][0]) + n * KSTRIDE;
    par ^= 1;
#else
    const float* krow = Kb + (size_t)((t << 4) + n) * DD;
    if (t + NWAVES < NTILES)
      __builtin_prefetch(Kb + (size_t)(((t + NWAVES) << 4) + n) * DD, 0, 1);
#endif

    // A fragment (16-bit A 16x32 layout): element j ->
    //   kk = (j>=8 ? 16 : 0) + h*8 + (j&7), per slice of 32 dims.
    const float4* k4 = (const float4*)(krow + (h << 3));  // base dim = h*8
    const float4 c0 = k4[0],  c1 = k4[1];    // dims h*8    .. h*8+7
    const float4 c2 = k4[4],  c3 = k4[5];    // dims 16+h*8 ..
    const float4 c4 = k4[8],  c5 = k4[9];    // dims 32+h*8 ..
    const float4 c6 = k4[12], c7 = k4[13];   // dims 48+h*8 ..

    v16bf ak0, ak1;
    ak0[0]  = (__bf16)c0.x; ak0[1]  = (__bf16)c0.y; ak0[2]  = (__bf16)c0.z; ak0[3]  = (__bf16)c0.w;
    ak0[4]  = (__bf16)c1.x; ak0[5]  = (__bf16)c1.y; ak0[6]  = (__bf16)c1.z; ak0[7]  = (__bf16)c1.w;
    ak0[8]  = (__bf16)c2.x; ak0[9]  = (__bf16)c2.y; ak0[10] = (__bf16)c2.z; ak0[11] = (__bf16)c2.w;
    ak0[12] = (__bf16)c3.x; ak0[13] = (__bf16)c3.y; ak0[14] = (__bf16)c3.z; ak0[15] = (__bf16)c3.w;
    ak1[0]  = (__bf16)c4.x; ak1[1]  = (__bf16)c4.y; ak1[2]  = (__bf16)c4.z; ak1[3]  = (__bf16)c4.w;
    ak1[4]  = (__bf16)c5.x; ak1[5]  = (__bf16)c5.y; ak1[6]  = (__bf16)c5.z; ak1[7]  = (__bf16)c5.w;
    ak1[8]  = (__bf16)c6.x; ak1[9]  = (__bf16)c6.y; ak1[10] = (__bf16)c6.z; ak1[11] = (__bf16)c6.w;
    ak1[12] = (__bf16)c7.x; ak1[13] = (__bf16)c7.y; ak1[14] = (__bf16)c7.z; ak1[15] = (__bf16)c7.w;

    v8f acc = {};
    acc = __builtin_amdgcn_wmma_f32_16x16x32_bf16(false, ak0, false, bq0,
                                                  (short)0, acc, false, false);
    acc = __builtin_amdgcn_wmma_f32_16x16x32_bf16(false, ak1, false, bq1,
                                                  (short)0, acc, false, false);

    // Per-query tile stats: 8 local keys reduce in-lane, 1 cross-half swap.
    float lv[8];
    #pragma unroll
    for (int r = 0; r < 8; ++r) lv[r] = acc[r] * SCALE;

    float tl = lv[0];
    #pragma unroll
    for (int r = 1; r < 8; ++r) tl = fmaxf(tl, lv[r]);
    const float tmax = fmaxf(tl, xhalf(tl));           // tile max (this query)

    float ps = 0.0f;
    #pragma unroll
    for (int r = 0; r < 8; ++r) ps += __expf(lv[r] - tmax);
    const float ts = ps + xhalf(ps);                   // tile expsum

    if (h == 0) {
      tileMax[n][t] = tmax;
      tileSum[n][t] = ts;
    }

    const float nm = fmaxf(m_run, tmax);
    l_run = l_run * __expf(m_run - nm) + ts * __expf(tmax - nm);
    m_run = nm;
  }

  if (h == 0) {
    wpM[wave][n] = m_run;
    wpL[wave][n] = l_run;
  }
  __syncthreads();

  // ---- combine wave partials into per-query softmax stats ----
  if (tid < QT) {
    float M = -INFINITY;
    #pragma unroll
    for (int w = 0; w < NWAVES; ++w) M = fmaxf(M, wpM[w][tid]);
    float L = 0.0f;
    #pragma unroll
    for (int w = 0; w < NWAVES; ++w) L += wpL[w][tid] * __expf(wpM[w][tid] - M);
    rowM[tid] = M;
    rowL[tid] = L;
  }
  __syncthreads();

  // ---- rescale per-tile sums to a common base (CDF table) ----
  for (int i = tid; i < QT * NTILES; i += 128) {
    const int row = i >> 8;
    const int t   = i & (NTILES - 1);
    tileSum[row][t] *= __expf(tileMax[row][t] - rowM[row]);
  }
  __syncthreads();

  // ---- multinomial sampling: inverse CDF over tile sums, scalar refine ----
  for (int task = tid; task < QT * SS; task += 128) {
    const int row = task >> 5;      // query within tile
    const int s   = task & 31;
    const unsigned hsh =
        hash32(((unsigned)(b * NQ + q0 + row) << 5) ^ (unsigned)s ^ 0x2A2A2A2Au);
    const float u = ((float)(hsh & 0xFFFFFFu) + 0.5f) * (1.0f / 16777216.0f);
    const float target = u * rowL[row];

    float acc = 0.0f;
    int t = NTILES - 1;
    for (int i = 0; i < NTILES; ++i) {
      const float ns = acc + tileSum[row][i];
      if (ns >= target) { t = i; break; }
      acc = ns;
    }

    int chosen = (t << 4) + 15;
    const float* qrow = &Qs[row * DD];
    for (int kk = 0; kk < 16; ++kk) {
      const int key = (t << 4) + kk;
      const float* krow = Kb + (size_t)key * DD;
      float dot = 0.0f;
      #pragma unroll 8
      for (int d = 0; d < DD; ++d) dot = fmaf(qrow[d], krow[d], dot);
      acc += __expf(dot * SCALE - rowM[row]);
      if (acc >= target) { chosen = key; break; }
    }
    sampKey[row][s] = (unsigned short)chosen;
  }
  __syncthreads();

  // ---- output: average the 32 sampled V rows (deterministic gather) ----
  for (int i = tid; i < QT * DD; i += 128) {
    const int row = i >> 6;
    const int d   = i & (DD - 1);
    float sum = 0.0f;
    #pragma unroll 4
    for (int s = 0; s < SS; ++s)
      sum += Vb[(size_t)sampKey[row][s] * DD + d];
    Out[(size_t)(b * NQ + q0 + row) * DD + d] = sum * (1.0f / SS);
  }
}

extern "C" void kernel_launch(void* const* d_in, const int* in_sizes, int n_in,
                              void* d_out, int out_size, void* d_ws, size_t ws_size,
                              hipStream_t stream) {
  (void)in_sizes; (void)n_in; (void)out_size; (void)d_ws; (void)ws_size;
  const float* Q = (const float*)d_in[0];
  const float* K = (const float*)d_in[1];
  const float* V = (const float*)d_in[2];
  float* Out = (float*)d_out;

  dim3 grid(B_ * (NQ / QT));  // 512 workgroups: one per (batch, 16-query tile)
  dim3 block(128);            // 4 wave32s
  qattn_sample_kernel<<<grid, block, 0, stream>>>(Q, K, V, Out);
}
